// MoSEModel_26482768347312
// MI455X (gfx1250) — matrix operand order repaired and checked by reading, not compiled, not run on hardware
//
#include <hip/hip_runtime.h>
#include <stdint.h>

// ---------------- problem constants ----------------
#define B_      64
#define S_      512
#define H_      256
#define E_      5
#define NH_     3
#define MLP_    384
#define GATES_  1024                       // 4*H
#define NPASS_  12                         // 2 shared + 5 experts * 2 layers
#define TILE_HALFS_ 4096                   // one (16-row, t) tile = 8 frags * 512 halfs
static const size_t PASS_HALFS_ = (size_t)1024 * 512;       // frags per pass * 512
static const size_t SEQ_HALFS_  = (size_t)B_ * S_ * H_;     // one sequence buffer

// ---------------- WMMA types ----------------
typedef __bf16 v16bf __attribute__((ext_vector_type(16)));
typedef float  v8f   __attribute__((ext_vector_type(8)));
union Frag { uint4 u[2]; v16bf v; };

__device__ __forceinline__ unsigned short f2bf(float f) {
  unsigned u = __float_as_uint(f);
  u += 0x7fffu + ((u >> 16) & 1u);          // round-to-nearest-even
  return (unsigned short)(u >> 16);
}
__device__ __forceinline__ float bf2f(unsigned short h) {
  return __uint_as_float(((unsigned)h) << 16);
}
__device__ __forceinline__ float sigf(float x) { return 1.f / (1.f + __expf(-x)); }

// Tile-local half-offset of element (row m, col j) in A-fragment sequence layout.
// 16-bit A 16x32 (ISA 7.12.2): lane = (half*16 + m), within-lane half index = v*2+p,
// K = (v<4 ? 2v : 16+2(v-4)) + 8*half + p.  kc = j/32 selects the fragment.
__device__ __forceinline__ int swz(int m, int j) {
  int kc = j >> 5, kl = j & 31;
  int hi = (kl >> 4) & 1, r = kl & 15;
  int half = r >> 3, q = r & 7;
  int v = hi * 4 + (q >> 1), p = q & 1;
  return kc * 512 + (half * 16 + m) * 16 + v * 2 + p;
}

// ---------------- prep: x -> bf16 A-frag layout + mask ----------------
__global__ __launch_bounds__(256) void prep_x_kernel(const float* __restrict__ x,
    unsigned short* __restrict__ xswz, float* __restrict__ maskf) {
  int bid = blockIdx.x; int mb = bid >> 9; int t = bid & 511;
  int tid = threadIdx.x; int m = tid >> 4, jb = tid & 15;
  __shared__ float sP[16][16];
  int b = mb * 16 + m;
  const float* xp = x + ((size_t)b * S_ + t) * H_ + jb * 16;
  unsigned short* op = xswz + (size_t)(mb * S_ + t) * TILE_HALFS_;
  float s = 0.f;
#pragma unroll
  for (int jj = 0; jj < 16; jj++) { float v = xp[jj]; s += v; op[swz(m, jb * 16 + jj)] = f2bf(v); }
  sP[m][jb] = s; __syncthreads();
  if (jb == 0) {
    float a = 0; for (int k = 0; k < 16; k++) a += sP[m][k];
    maskf[(size_t)b * S_ + t] = (a != 0.f) ? 1.f : 0.f;
  }
}

// ---------------- prep: weights -> combined [Wih;Whh] bf16 B-frag layout ----------------
// B fragment (32x16): lane = half*16 + N, half-index hh: K = (hh<8 ? hh : 8+hh) + 8*half.
__global__ __launch_bounds__(256) void prep_w_kernel(
    const float* __restrict__ rWih, const float* __restrict__ rWhh,
    const float* __restrict__ eWih, const float* __restrict__ eWhh,
    unsigned short* __restrict__ wswz) {
  int idx = blockIdx.x * 256 + threadIdx.x;        // 12*1024 frags * 32 lanes
  int fig = idx >> 5, lane = idx & 31;
  int pass = fig >> 10, fi = fig & 1023;
  int nt = fi >> 4, kc = fi & 15;
  int kb = kc * 32 + (lane >> 4) * 8;              // fused K: [0,256)=Wih, [256,512)=Whh
  unsigned short* dst = wswz + (size_t)fig * 512 + lane * 16;
#pragma unroll
  for (int hh = 0; hh < 16; hh++) {
    int n = nt * 16 + (lane & 15); int k = kb + (hh < 8 ? hh : 8 + hh);
    float v;
    if (pass < 2) {
      size_t base = ((size_t)pass * GATES_ + n) * H_;
      v = (k < H_) ? rWih[base + k] : rWhh[base + (k - H_)];
    } else {
      int q = pass - 2;                            // (e*2 + l) flattened
      size_t base = ((size_t)q * GATES_ + n) * H_;
      v = (k < H_) ? eWih[base + k] : eWhh[base + (k - H_)];
    }
    dst[hh] = f2bf(v);
  }
}

__global__ void prep_bias_kernel(const float* __restrict__ rbih, const float* __restrict__ rbhh,
    const float* __restrict__ ebih, const float* __restrict__ ebhh, float* __restrict__ bias) {
  int idx = blockIdx.x * 256 + threadIdx.x;
  if (idx >= NPASS_ * GATES_) return;
  int pass = idx >> 10, n = idx & 1023;
  float v;
  if (pass < 2) v = rbih[pass * GATES_ + n] + rbhh[pass * GATES_ + n];
  else { int q = pass - 2; v = ebih[q * GATES_ + n] + ebhh[q * GATES_ + n]; }
  bias[idx] = v;
}

// ---------------- gate path: mask-attn over x + per-head MLP + select D ----------------
__global__ __launch_bounds__(256) void gate_kernel(
    const float* __restrict__ x, const float* __restrict__ attn_w,
    const float* __restrict__ W1, const float* __restrict__ b1,
    const float* __restrict__ W2, const float* __restrict__ b2,
    const int* __restrict__ D, float* __restrict__ gate_value) {
  int b = blockIdx.x, tid = threadIdx.x;
  __shared__ float sAw[256], sE[512], sGf[256], sH1[NH_ * MLP_], sR[256], sLg[NH_ * E_];
  __shared__ float sTot;
  sAw[tid] = attn_w[tid];
  __syncthreads();
  for (int tt = tid; tt < S_; tt += 256) {
    const float* xp = x + ((size_t)b * S_ + tt) * H_;
    float s = 0, sx = 0;
    for (int j = 0; j < H_; j++) { float v = xp[j]; s += v * sAw[j]; sx += v; }
    sE[tt] = (sx != 0.f) ? s : -1e30f;
  }
  __syncthreads();
  float mx = fmaxf(sE[tid], sE[tid + 256]);
  sR[tid] = mx; __syncthreads();
  for (int o = 128; o > 0; o >>= 1) { if (tid < o) sR[tid] = fmaxf(sR[tid], sR[tid + o]); __syncthreads(); }
  mx = sR[0]; __syncthreads();
  float e0 = __expf(sE[tid] - mx), e1 = __expf(sE[tid + 256] - mx);
  sE[tid] = e0; sE[tid + 256] = e1; sR[tid] = e0 + e1; __syncthreads();
  for (int o = 128; o > 0; o >>= 1) { if (tid < o) sR[tid] += sR[tid + o]; __syncthreads(); }
  if (tid == 0) sTot = sR[0];
  __syncthreads();
  { float a = 0; const float* xc = x + (size_t)b * S_ * H_ + tid;
    for (int tt = 0; tt < S_; tt++) a += sE[tt] * xc[(size_t)tt * H_];
    sGf[tid] = a / sTot; }
  __syncthreads();
  for (int i = tid; i < NH_ * MLP_; i += 256) {
    int h = i / MLP_, mm = i - h * MLP_;
    const float* wp = W1 + ((size_t)h * MLP_ + mm) * H_;
    float s = 0; for (int j = 0; j < H_; j++) s += wp[j] * sGf[j];
    s += b1[h * MLP_ + mm]; sH1[i] = s > 0.f ? s : 0.f;
  }
  __syncthreads();
  if (tid < NH_ * E_) {
    int h = tid / E_, k = tid - h * E_;
    const float* wp = W2 + ((size_t)(h * E_ + k)) * MLP_;
    const float* hp = sH1 + h * MLP_;
    float s = 0; for (int mm = 0; mm < MLP_; mm++) s += wp[mm] * hp[mm];
    sLg[tid] = s + b2[h * E_ + k];
  }
  __syncthreads();
  if (tid == 0) {
    int d = D[b];
    float m2 = -1e30f; for (int k = 0; k < E_; k++) m2 = fmaxf(m2, sLg[d * E_ + k]);
    float z = 0.f, ee[E_];
    for (int k = 0; k < E_; k++) { ee[k] = __expf(sLg[d * E_ + k] - m2); z += ee[k]; }
    for (int k = 0; k < E_; k++) gate_value[b * E_ + k] = ee[k] / z;
  }
}

// ---------------- LayerNorm over hidden dim (frag layout in/out) ----------------
__global__ __launch_bounds__(256) void ln_kernel(const unsigned short* __restrict__ in,
    const float* __restrict__ lnw, const float* __restrict__ lnb,
    unsigned short* __restrict__ out) {
  int bid = blockIdx.x; int mb = bid >> 9, t = bid & 511;
  int tid = threadIdx.x, m = tid >> 4, jb = tid & 15;
  __shared__ float sS[16][16], sQ[16][16], sMu[16], sIv[16];
  size_t tile = (size_t)(mb * S_ + t) * TILE_HALFS_;
  float v[16]; float s = 0, q = 0;
#pragma unroll
  for (int jj = 0; jj < 16; jj++) {
    float f = bf2f(in[tile + swz(m, jb * 16 + jj)]);
    v[jj] = f; s += f; q += f * f;
  }
  sS[m][jb] = s; sQ[m][jb] = q; __syncthreads();
  if (jb == 0) {
    float a = 0, c = 0; for (int k = 0; k < 16; k++) { a += sS[m][k]; c += sQ[m][k]; }
    float mu = a / H_; float var = c / H_ - mu * mu;
    sMu[m] = mu; sIv[m] = rsqrtf(var + 1e-5f);
  }
  __syncthreads();
  float mu = sMu[m], iv = sIv[m];
#pragma unroll
  for (int jj = 0; jj < 16; jj++) {
    int j = jb * 16 + jj;
    float y = (v[jj] - mu) * iv * lnw[j] + lnb[j];
    out[tile + swz(m, j)] = f2bf(y);
  }
}

// ---------------- fused LSTM layer pass (persistent, WMMA recurrence) ----------------
// grid.x = batch tile (16 rows), grid.y = expert.
// Per step: gates[16,1024] = [x_t | h_{t-1}] (16x512 bf16) @ Wfused^T via 1024 wmma_bf16 (128 per wave).
// Input tiles are DMA'd into double-buffered LDS with global_load_async_to_lds_b128,
// one step ahead (ASYNCcnt pipelined: wait<=2 keeps next tile's loads in flight).
// Wave w owns cols [32w,32w+32) of each gate i/f/g/o -> c stays entirely in VGPRs.
// Optional fused online-softmax attention (expert layer 1).
__global__ __launch_bounds__(256) void lstm_pass_kernel(
    const unsigned short* __restrict__ in_seq, size_t in_stride_e,
    const unsigned short* __restrict__ W, size_t w_stride_e,
    const float* __restrict__ bias, size_t b_stride_e,
    unsigned short* __restrict__ out_seq, size_t out_stride_e,
    const float* __restrict__ mask_w, const float* __restrict__ maskf,
    float* __restrict__ rep_e) {
  int mb = blockIdx.x, e = blockIdx.y;
  int tid = threadIdx.x, lane = tid & 31, w = tid >> 5;
  int rowh = lane >> 4, col = lane & 15;
  int jb0 = w * 32;
  __shared__ __align__(16) unsigned short sH[TILE_HALFS_];      // h_t, A-frag layout (8KB)
  __shared__ __align__(16) unsigned short sIn[2][TILE_HALFS_];  // async-staged input tiles
  __shared__ float sA[16 * H_];                                 // online attn accumulator
  __shared__ float sMW[H_];
  __shared__ float sPart[16][16];
  __shared__ float sM[16], sZ[16], sScl[16], sPp[16];

  const unsigned short* in_e = in_seq + (size_t)e * in_stride_e;
  const unsigned short* We = W + (size_t)e * w_stride_e;
  const float* be = bias + (size_t)e * b_stride_e;
  unsigned short* out_e = out_seq ? (out_seq + (size_t)e * out_stride_e) : (unsigned short*)0;
  const bool attn = (mask_w != 0);

  for (int i = tid; i < TILE_HALFS_; i += 256) sH[i] = 0;
  if (attn) {
    const float* mw = mask_w + (size_t)e * H_;
    sMW[tid] = mw[tid];
    if (tid < 16) { sM[tid] = -1e30f; sZ[tid] = 0.f; }
    for (int i = tid; i < 16 * H_; i += 256) sA[i] = 0.f;
  }
  float bb[4][2];
#pragma unroll
  for (int g = 0; g < 4; g++)
#pragma unroll
    for (int ct = 0; ct < 2; ct++)
      bb[g][ct] = be[g * 256 + jb0 + ct * 16 + col];
  float cr[2][8];
#pragma unroll
  for (int ct = 0; ct < 2; ct++)
#pragma unroll
    for (int r = 0; r < 8; r++) cr[ct][r] = 0.f;

  // async DMA of tile 0 into sIn[0] (each thread covers 32 contiguous bytes)
  {
    unsigned ldsa = (unsigned)(size_t)&sIn[0][tid * 16];
    const unsigned short* g = in_e + (size_t)(mb * S_) * TILE_HALFS_ + tid * 16;
    asm volatile(
        "global_load_async_to_lds_b128 %0, %1, off\n\t"
        "global_load_async_to_lds_b128 %0, %1, off offset:16"
        :: "v"(ldsa), "v"(g));
  }
  __syncthreads();

  for (int t = 0; t < S_; ++t) {
    int cur = t & 1;
    // kick off DMA of the next input tile into the other buffer
    if (t + 1 < S_) {
      unsigned ldsa = (unsigned)(size_t)&sIn[cur ^ 1][tid * 16];
      const unsigned short* g = in_e + (size_t)(mb * S_ + t + 1) * TILE_HALFS_ + tid * 16;
      asm volatile(
          "global_load_async_to_lds_b128 %0, %1, off\n\t"
          "global_load_async_to_lds_b128 %0, %1, off offset:16"
          :: "v"(ldsa), "v"(g));
    }
    // Opaque zero offset, redefined each iteration: blocks LICM from hoisting
    // (and spilling) the 2048 loop-invariant weight-fragment loads, while the
    // GEP stays rooted at the kernarg pointer so address-space inference still
    // selects global_load (pure LOADcnt) instead of flat_load (LOADcnt+DScnt).
    size_t zoff = 0;
    asm volatile("" : "+s"(zoff));
    const unsigned short* Wt = We + zoff;

    v8f acc[4][2];
    { v8f z = {0.f, 0.f, 0.f, 0.f, 0.f, 0.f, 0.f, 0.f};
#pragma unroll
      for (int g = 0; g < 4; g++) { acc[g][0] = z; acc[g][1] = z; } }

    // ---- recurrent half first: K chunks 8..15 read h_{t-1} from LDS ----
#pragma unroll
    for (int kc = 0; kc < 8; kc++) {
      Frag a;
      const uint4* p = (const uint4*)&sH[kc * 512 + lane * 16];
      a.u[0] = p[0]; a.u[1] = p[1];
#pragma unroll
      for (int g = 0; g < 4; g++) {
#pragma unroll
        for (int ct = 0; ct < 2; ct++) {
          const uint4* bp = (const uint4*)(Wt +
              (size_t)((g * 16 + w * 2 + ct) * 16 + 8 + kc) * 512 + lane * 16);
          Frag bf_; bf_.u[0] = bp[0]; bf_.u[1] = bp[1];
          acc[g][ct] = __builtin_amdgcn_wmma_f32_16x16x32_bf16(
              false, a.v, false, bf_.v, (short)0, acc[g][ct], false, false);
        }
      }
    }
    // drain the 2 oldest async loads (tile t); tile t+1's stay in flight
    if (t + 1 < S_) asm volatile("s_wait_asynccnt 0x2" ::: "memory");
    else            asm volatile("s_wait_asynccnt 0x0" ::: "memory");
    __syncthreads();   // all sH reads done; sIn[cur] complete and visible

    // ---- input half: K chunks 0..7 read staged x_t from LDS ----
#pragma unroll
    for (int kc = 0; kc < 8; kc++) {
      Frag a;
      const uint4* p = (const uint4*)&sIn[cur][kc * 512 + lane * 16];
      a.u[0] = p[0]; a.u[1] = p[1];
#pragma unroll
      for (int g = 0; g < 4; g++) {
#pragma unroll
        for (int ct = 0; ct < 2; ct++) {
          const uint4* bp = (const uint4*)(Wt +
              (size_t)((g * 16 + w * 2 + ct) * 16 + kc) * 512 + lane * 16);
          Frag bf_; bf_.u[0] = bp[0]; bf_.u[1] = bp[1];
          acc[g][ct] = __builtin_amdgcn_wmma_f32_16x16x32_bf16(
              false, a.v, false, bf_.v, (short)0, acc[g][ct], false, false);
        }
      }
    }

    // gates -> (c,h) entirely in registers; write h (bf16, swizzled) to LDS (+global)
    unsigned short* outT = out_e ? (out_e + (size_t)(mb * S_ + t) * TILE_HALFS_) : (unsigned short*)0;
#pragma unroll
    for (int ct = 0; ct < 2; ct++) {
#pragma unroll
      for (int r = 0; r < 8; r++) {
        float gi  = acc[0][ct][r] + bb[0][ct];
        float gf_ = acc[1][ct][r] + bb[1][ct];
        float gg  = acc[2][ct][r] + bb[2][ct];
        float go  = acc[3][ct][r] + bb[3][ct];
        float cn = sigf(gf_) * cr[ct][r] + sigf(gi) * tanhf(gg);
        cr[ct][r] = cn;
        float hv = sigf(go) * tanhf(cn);
        int m = r + rowh * 8;
        int j = jb0 + ct * 16 + col;
        int off = swz(m, j);
        unsigned short hb = f2bf(hv);
        sH[off] = hb;
        if (outT) outT[off] = hb;
      }
    }
    __syncthreads();

    if (attn) {  // streaming (online-softmax) masked attention over h_t
      int m = tid >> 4, jb = tid & 15;
      float hv[16]; float part = 0.f;
#pragma unroll
      for (int jj = 0; jj < 16; jj++) {
        float f = bf2f(sH[swz(m, jb * 16 + jj)]);
        hv[jj] = f; part += f * sMW[jb * 16 + jj];
      }
      sPart[m][jb] = part;
      __syncthreads();
      if (tid < 16) {
        float s = 0; for (int k = 0; k < 16; k++) s += sPart[tid][k];
        float mval = maskf[(size_t)(mb * 16 + tid) * S_ + t];
        if (mval != 0.f) {
          float Mo = sM[tid]; float Mn = fmaxf(Mo, s);
          float scl = __expf(Mo - Mn); float pp = __expf(s - Mn);
          sZ[tid] = sZ[tid] * scl + pp; sM[tid] = Mn;
          sScl[tid] = scl; sPp[tid] = pp;
        } else { sScl[tid] = 1.f; sPp[tid] = 0.f; }
      }
      __syncthreads();
      float scl = sScl[m], pp = sPp[m];
#pragma unroll
      for (int jj = 0; jj < 16; jj++) {
        int idx = m * H_ + jb * 16 + jj;
        sA[idx] = sA[idx] * scl + pp * hv[jj];
      }
    }
  }

  if (attn) {
    __syncthreads();
    int m = tid >> 4, jb = tid & 15;
    float z = sZ[m]; float inv = (z > 0.f) ? 1.f / z : 0.f;
#pragma unroll
    for (int jj = 0; jj < 16; jj++)
      rep_e[(((size_t)e * B_) + (mb * 16 + m)) * H_ + jb * 16 + jj] =
          sA[m * H_ + jb * 16 + jj] * inv;
  }
}

// ---------------- mixture + per-sample head + softmax ----------------
__global__ __launch_bounds__(256) void final_kernel(
    const float* __restrict__ gv, const float* __restrict__ rep_e,
    const float* __restrict__ hW, const float* __restrict__ hb,
    const int* __restrict__ D, float* __restrict__ out) {
  int b = blockIdx.x, tid = threadIdx.x;
  __shared__ float sR0[256], sR1[256];
  float r = 0.f;
  for (int e = 0; e < E_; e++)
    r += gv[b * E_ + e] * rep_e[(((size_t)e * B_) + b) * H_ + tid];
  int d = D[b];
  sR0[tid] = r * hW[((size_t)d * 2 + 0) * H_ + tid];
  sR1[tid] = r * hW[((size_t)d * 2 + 1) * H_ + tid];
  __syncthreads();
  for (int o = 128; o > 0; o >>= 1) {
    if (tid < o) { sR0[tid] += sR0[tid + o]; sR1[tid] += sR1[tid + o]; }
    __syncthreads();
  }
  if (tid == 0) {
    float l0 = sR0[0] + hb[d * 2], l1 = sR1[0] + hb[d * 2 + 1];
    float mx = fmaxf(l0, l1);
    float e0 = __expf(l0 - mx), e1 = __expf(l1 - mx);
    out[b * 2] = e0 / (e0 + e1);
    out[b * 2 + 1] = e1 / (e0 + e1);
  }
}

// ---------------- host orchestration ----------------
extern "C" void kernel_launch(void* const* d_in, const int* in_sizes, int n_in,
                              void* d_out, int out_size, void* d_ws, size_t ws_size,
                              hipStream_t stream) {
  const float* x      = (const float*)d_in[0];
  const float* attn_w = (const float*)d_in[1];
  const float* rWih   = (const float*)d_in[2];
  const float* rWhh   = (const float*)d_in[3];
  const float* rbih   = (const float*)d_in[4];
  const float* rbhh   = (const float*)d_in[5];
  const float* lnw    = (const float*)d_in[6];
  const float* lnb    = (const float*)d_in[7];
  const float* eWih   = (const float*)d_in[8];
  const float* eWhh   = (const float*)d_in[9];
  const float* ebih   = (const float*)d_in[10];
  const float* ebhh   = (const float*)d_in[11];
  const float* mask_w = (const float*)d_in[12];
  const float* gW1    = (const float*)d_in[13];
  const float* gb1    = (const float*)d_in[14];
  const float* gW2    = (const float*)d_in[15];
  const float* gb2    = (const float*)d_in[16];
  const float* hW     = (const float*)d_in[17];
  const float* hb     = (const float*)d_in[18];
  const int*   D      = (const int*)d_in[19];
  float* out = (float*)d_out;
  (void)in_sizes; (void)n_in; (void)out_size; (void)ws_size;

  char* ws = (char*)d_ws;
  size_t off = 0;
  auto alloc = [&](size_t bytes) { size_t o = off; off += (bytes + 255) & ~(size_t)255; return o; };
  float*          maskf = (float*)(ws + alloc((size_t)B_ * S_ * 4));
  unsigned short* xswz  = (unsigned short*)(ws + alloc(SEQ_HALFS_ * 2));
  unsigned short* seqA  = (unsigned short*)(ws + alloc(SEQ_HALFS_ * 2));
  unsigned short* seqB  = (unsigned short*)(ws + alloc(SEQ_HALFS_ * 2));
  unsigned short* ebuf  = (unsigned short*)(ws + alloc((size_t)E_ * SEQ_HALFS_ * 2));
  unsigned short* wswz  = (unsigned short*)(ws + alloc((size_t)NPASS_ * PASS_HALFS_ * 2));
  float*          biasw = (float*)(ws + alloc((size_t)NPASS_ * GATES_ * 4));
  float*          gvw   = (float*)(ws + alloc((size_t)B_ * E_ * 4));
  float*          repe  = (float*)(ws + alloc((size_t)E_ * B_ * H_ * 4));

  prep_x_kernel<<<dim3((B_ / 16) * S_), 256, 0, stream>>>(x, xswz, maskf);
  prep_w_kernel<<<dim3((NPASS_ * 1024 * 32) / 256), 256, 0, stream>>>(rWih, rWhh, eWih, eWhh, wswz);
  prep_bias_kernel<<<dim3((NPASS_ * GATES_ + 255) / 256), 256, 0, stream>>>(rbih, rbhh, ebih, ebhh, biasw);
  gate_kernel<<<dim3(B_), 256, 0, stream>>>(x, attn_w, gW1, gb1, gW2, gb2, D, gvw);

  // shared L0: x -> seqA
  lstm_pass_kernel<<<dim3(4, 1), 256, 0, stream>>>(xswz, 0, wswz + 0 * PASS_HALFS_, 0,
      biasw + 0 * GATES_, 0, seqA, 0, nullptr, maskf, nullptr);
  // shared L1: seqA -> seqB
  lstm_pass_kernel<<<dim3(4, 1), 256, 0, stream>>>(seqA, 0, wswz + 1 * PASS_HALFS_, 0,
      biasw + 1 * GATES_, 0, seqB, 0, nullptr, maskf, nullptr);
  // LayerNorm: seqB -> seqA (feature)
  ln_kernel<<<dim3((B_ / 16) * S_), 256, 0, stream>>>(seqB, lnw, lnb, seqA);
  // experts L0 (5 experts in parallel): feature -> ebuf[e]
  lstm_pass_kernel<<<dim3(4, E_), 256, 0, stream>>>(seqA, 0, wswz + 2 * PASS_HALFS_,
      2 * PASS_HALFS_, biasw + 2 * GATES_, 2 * GATES_, ebuf, SEQ_HALFS_, nullptr, maskf, nullptr);
  // experts L1 + fused online-softmax attention -> rep_e
  lstm_pass_kernel<<<dim3(4, E_), 256, 0, stream>>>(ebuf, SEQ_HALFS_, wswz + 3 * PASS_HALFS_,
      2 * PASS_HALFS_, biasw + 3 * GATES_, 2 * GATES_, nullptr, 0, mask_w, maskf, repe);

  final_kernel<<<dim3(B_), 256, 0, stream>>>(gvw, repe, hW, hb, D, out);
}
